// MPGATConv_39840116638273
// MI455X (gfx1250) — matrix operand order repaired and checked by reading, not compiled, not run on hardware
//
#include <hip/hip_runtime.h>
#include <math.h>

#define N_NODES 50000
#define N_EDGES 800000
#define IN_F 128
#define EF_F 128
#define DD 32
#define HH 8
#define HD 256
#define CUTOFF_R 5.0f
#define NEG_SLOPE 0.2f
#define LOG2F_ 0.69314718055994530942f

typedef __attribute__((ext_vector_type(16))) _Float16 v16h;
typedef __attribute__((ext_vector_type(8)))  float    v8f;

__device__ __forceinline__ float ssp(float x) {
    // shifted softplus: softplus(x) - log(2), overflow-safe
    float sp = (x > 20.f) ? x : log1pf(__expf(x));
    return sp - LOG2F_;
}
__device__ __forceinline__ float coscut(float r) {
    return (r < CUTOFF_R) ? 0.5f * (__cosf(3.14159265358979323846f * r / CUTOFF_R) + 1.f) : 0.f;
}
__device__ __forceinline__ void atomicMaxF(float* addr, float v) {
    // sign-split trick: correct total-order float max via integer atomics
    if (v >= 0.f) atomicMax((int*)addr, __float_as_int(v));
    else          atomicMin((unsigned int*)addr, (unsigned int)__float_as_int(v));
}

// ---------------- init: rst = bias, emax = -inf, esum = 0 ----------------
__global__ void init_kernel(float* __restrict__ rst, const float* __restrict__ bias,
                            float* __restrict__ emax, float* __restrict__ esum) {
    int i = blockIdx.x * 256 + threadIdx.x;          // grid covers N*256 exactly
    rst[i] = bias[i & 255];
    if (i < N_NODES * HH) { emax[i] = -INFINITY; esum[i] = 0.f; }
}

// ---------------- feat = nfeat @ W_fc^T  (WMMA, one 16x16 tile per wave) ----------------
__global__ void feat_wmma_kernel(const float* __restrict__ nfeat,
                                 const float* __restrict__ Wfc,
                                 float* __restrict__ feat) {
    const int lane = threadIdx.x & 31;
    const int wave = threadIdx.x >> 5;
    const int col0 = (blockIdx.x * 4 + wave) * 16;   // 16 column tiles of 256
    const int row0 = blockIdx.y * 16;                // node tile
    const int ml   = lane & 15;
    const int ka   = (lane < 16) ? 0 : 8;            // unused; A loads contiguous k = 16h + s via kb, like B
    const int kb   = (lane < 16) ? 0 : 16;           // B-matrix K split
    v8f c = {};
#pragma unroll
    for (int kc = 0; kc < IN_F; kc += 32) {
        v16h a, b;
        const float* ap = nfeat + (row0 + ml) * IN_F + kc;
#pragma unroll
        for (int i = 0; i < 16; ++i) a[i] = (_Float16)ap[kb + i];
        const float* bp = Wfc + (col0 + ml) * IN_F + kc + kb;   // B[k][n] = Wfc[n][k]
#pragma unroll
        for (int i = 0; i < 16; ++i) b[i] = (_Float16)bp[i];
        c = __builtin_amdgcn_wmma_f32_16x16x32_f16(false, a, false, b, (short)0, c, false, false);
    }
    const int rbase = row0 + ((lane < 16) ? 0 : 8);
#pragma unroll
    for (int v = 0; v < 8; ++v) feat[(rbase + v) * HD + (col0 + ml)] = c[v];
}

// ---------------- el/er = <feat, attn_l/r> per (node, head) ----------------
__global__ void node_attn_kernel(const float* __restrict__ feat,
                                 const float* __restrict__ attn_l, const float* __restrict__ attn_r,
                                 float* __restrict__ el, float* __restrict__ er) {
    int i = blockIdx.x * 256 + threadIdx.x;
    if (i >= N_NODES * HH) return;
    int n = i >> 3, h = i & 7;
    const float* f  = feat + n * HD + h * DD;
    const float* al = attn_l + h * DD;
    const float* ar = attn_r + h * DD;
    float sl = 0.f, sr = 0.f;
#pragma unroll
    for (int d = 0; d < DD; ++d) { float x = f[d]; sl += x * al[d]; sr += x * ar[d]; }
    el[i] = sl; er[i] = sr;
}

// ---------------- fused edge MLP -> ee[E,H] only (never materializes ef2) ----------------
// efeat tile is bulk-staged into LDS with async copies (ASYNCcnt), then fed to WMMA.
__global__ void edge_wmma_kernel(const float* __restrict__ efeat,
                                 const float* __restrict__ We1, const float* __restrict__ be1,
                                 const float* __restrict__ We2, const float* __restrict__ be2,
                                 const float* __restrict__ attn_e, const float* __restrict__ rij,
                                 float* __restrict__ ee) {
    __shared__ float etile[4][16][132];              // async-staged efeat tile, padded stride
    __shared__ float lds1[4][16][33];                // stage-1 output, padded stride
    const int lane = threadIdx.x & 31;
    const int wave = threadIdx.x >> 5;
    const int e0   = (blockIdx.x * 4 + wave) * 16;   // 16 edges per wave
    const int ml   = lane & 15;
    const int ka   = (lane < 16) ? 0 : 8;
    const int kb   = (lane < 16) ? 0 : 16;

    // ---- async bulk copy: 16 rows x 512B, one global_load_async_to_lds_b128 per row
    {
        const float* grow = efeat + (size_t)e0 * EF_F;
#pragma unroll
        for (int r = 0; r < 16; ++r) {
            const float* gaddr = grow + r * EF_F + lane * 4;            // 16B per lane
            unsigned laddr = (unsigned)(uintptr_t)(const void*)&etile[wave][r][lane * 4];
            asm volatile("global_load_async_to_lds_b128 %0, %1, off"
                         :: "v"(laddr), "v"(gaddr) : "memory");
        }
        asm volatile("s_wait_asynccnt 0x0" ::: "memory");
    }

    v16h a;
    // stage 1: T1[16x32] = ssp(efeat_tile @ We1^T + b1), K=128
    {
        v8f c0 = {}, c1 = {};
#pragma unroll
        for (int kc = 0; kc < EF_F; kc += 32) {
            const float* ap = &etile[wave][ml][kc];
#pragma unroll
            for (int i = 0; i < 16; ++i) a[i] = (_Float16)ap[kb + i];
            v16h b0, b1;
            const float* bp0 = We1 + ml * EF_F + kc + kb;          // d = ml
            const float* bp1 = We1 + (16 + ml) * EF_F + kc + kb;   // d = 16+ml
#pragma unroll
            for (int i = 0; i < 16; ++i) { b0[i] = (_Float16)bp0[i]; b1[i] = (_Float16)bp1[i]; }
            c0 = __builtin_amdgcn_wmma_f32_16x16x32_f16(false, a, false, b0, (short)0, c0, false, false);
            c1 = __builtin_amdgcn_wmma_f32_16x16x32_f16(false, a, false, b1, (short)0, c1, false, false);
        }
        const int rb = (lane < 16) ? 0 : 8;
#pragma unroll
        for (int v = 0; v < 8; ++v) {
            lds1[wave][rb + v][ml]      = ssp(c0[v] + be1[ml]);
            lds1[wave][rb + v][16 + ml] = ssp(c1[v] + be1[16 + ml]);
        }
    }
    __syncthreads();
    // reload T1 in A layout (f16), K = D = 32
#pragma unroll
    for (int i = 0; i < 16; ++i) {
        a[i] = (_Float16)lds1[wave][ml][kb + i];
    }
    const int rbase = (lane < 16) ? 0 : 8;
    float cut[8];
#pragma unroll
    for (int v = 0; v < 8; ++v) cut[v] = coscut(rij[e0 + rbase + v]);
    // stage 2: per head, 2 column tiles of We2; fold bias/ssp/cutoff/attn_edge + reduce over D.
    // Fully unrolled: 16 independent K=32 WMMAs all reusing the same A fragment.
#pragma unroll
    for (int h = 0; h < HH; ++h) {
        float s[8];
#pragma unroll
        for (int v = 0; v < 8; ++v) s[v] = 0.f;
#pragma unroll
        for (int t = 0; t < 2; ++t) {
            const int col = h * DD + t * 16 + ml;
            v16h b;
            const float* bp = We2 + col * DD + kb;                 // B[k=d][n=col] = We2[col][d]
#pragma unroll
            for (int i = 0; i < 16; ++i) b[i] = (_Float16)bp[i];
            v8f c = {};
            c = __builtin_amdgcn_wmma_f32_16x16x32_f16(false, a, false, b, (short)0, c, false, false);
            const float ae = attn_e[col];
            const float bb = be2[col];
#pragma unroll
            for (int v = 0; v < 8; ++v) s[v] += ssp(c[v] + bb) * cut[v] * ae;
        }
        // butterfly over the 16 lanes that share a row set
#pragma unroll
        for (int off = 1; off < 16; off <<= 1) {
#pragma unroll
            for (int v = 0; v < 8; ++v) s[v] += __shfl_xor(s[v], off, 32);
        }
        if (ml == 0) {
#pragma unroll
            for (int v = 0; v < 8; ++v) ee[(e0 + rbase + v) * HH + h] = s[v];
        }
    }
}

// ---------------- e = leaky(el[src]+er[dst]+ee); segment max ----------------
__global__ void edge_score_kernel(const float* __restrict__ el, const float* __restrict__ er,
                                  const float* __restrict__ ee,
                                  const int* __restrict__ src, const int* __restrict__ dst,
                                  float* __restrict__ ebuf, float* __restrict__ emax) {
    int i = blockIdx.x * 256 + threadIdx.x;
    if (i >= N_EDGES * HH) return;
    int e = i >> 3, h = i & 7;
    int s = src[e], d = dst[e];
    float v = el[s * HH + h] + er[d * HH + h] + ee[i];
    v = fmaxf(v, NEG_SLOPE * v);                     // leaky_relu
    ebuf[i] = v;
    atomicMaxF(&emax[d * HH + h], v);
}

// ---------------- ex = exp(e - emax[dst]); segment sum (in place) ----------------
__global__ void edge_exp_kernel(const int* __restrict__ dst, const float* __restrict__ emax,
                                float* __restrict__ ebuf, float* __restrict__ esum) {
    int i = blockIdx.x * 256 + threadIdx.x;
    if (i >= N_EDGES * HH) return;
    int e = i >> 3, h = i & 7;
    int dh = dst[e] * HH + h;
    float ex = __expf(ebuf[i] - emax[dh]);
    ebuf[i] = ex;
    atomicAdd(&esum[dh], ex);
}

// ---------------- rst[dst] += feat[src] * (ex/esum[dst] * ee) ----------------
__global__ void scatter_kernel(const int* __restrict__ src, const int* __restrict__ dst,
                               const float* __restrict__ feat, const float* __restrict__ ebuf,
                               const float* __restrict__ esum, const float* __restrict__ ee,
                               float* __restrict__ rst) {
    int i = blockIdx.x * 256 + threadIdx.x;          // E*64 threads, 4 floats each
    int e  = i >> 6;
    int r  = i & 63;
    int h  = r >> 3;
    int d4 = (r & 7) * 4;
    int sn = src[e], dn = dst[e];
    float aa = ebuf[e * HH + h] / esum[dn * HH + h] * ee[e * HH + h];
    const float4 f = *(const float4*)(feat + sn * HD + h * DD + d4);
    float* o = rst + dn * HD + h * DD + d4;
    atomicAdd(o + 0, f.x * aa);
    atomicAdd(o + 1, f.y * aa);
    atomicAdd(o + 2, f.z * aa);
    atomicAdd(o + 3, f.w * aa);
}

extern "C" void kernel_launch(void* const* d_in, const int* in_sizes, int n_in,
                              void* d_out, int out_size, void* d_ws, size_t ws_size,
                              hipStream_t stream) {
    (void)in_sizes; (void)n_in; (void)out_size; (void)ws_size;
    const float* nfeat  = (const float*)d_in[0];
    const float* efeat  = (const float*)d_in[1];
    const float* rij    = (const float*)d_in[2];
    const int*   src    = (const int*)d_in[3];
    const int*   dst    = (const int*)d_in[4];
    const float* Wfc    = (const float*)d_in[5];
    const float* We1    = (const float*)d_in[6];
    const float* be1    = (const float*)d_in[7];
    const float* We2    = (const float*)d_in[8];
    const float* be2    = (const float*)d_in[9];
    const float* attn_l = (const float*)d_in[10];
    const float* attn_r = (const float*)d_in[11];
    const float* attn_e = (const float*)d_in[12];
    const float* bias   = (const float*)d_in[13];
    float* rst = (float*)d_out;

    float* ws   = (float*)d_ws;
    float* feat = ws; ws += (size_t)N_NODES * HD;   // 12.8M
    float* el   = ws; ws += (size_t)N_NODES * HH;   // 0.4M
    float* er   = ws; ws += (size_t)N_NODES * HH;
    float* ee   = ws; ws += (size_t)N_EDGES * HH;   // 6.4M
    float* ebuf = ws; ws += (size_t)N_EDGES * HH;
    float* emax = ws; ws += (size_t)N_NODES * HH;
    float* esum = ws; ws += (size_t)N_NODES * HH;

    init_kernel<<<N_NODES * HD / 256, 256, 0, stream>>>(rst, bias, emax, esum);
    feat_wmma_kernel<<<dim3(4, N_NODES / 16), 128, 0, stream>>>(nfeat, Wfc, feat);
    node_attn_kernel<<<(N_NODES * HH + 255) / 256, 256, 0, stream>>>(feat, attn_l, attn_r, el, er);
    edge_wmma_kernel<<<N_EDGES / 64, 128, 0, stream>>>(efeat, We1, be1, We2, be2, attn_e, rij, ee);
    edge_score_kernel<<<(N_EDGES * HH + 255) / 256, 256, 0, stream>>>(el, er, ee, src, dst, ebuf, emax);
    edge_exp_kernel<<<(N_EDGES * HH + 255) / 256, 256, 0, stream>>>(dst, emax, ebuf, esum);
    scatter_kernel<<<N_EDGES * 64 / 256, 256, 0, stream>>>(src, dst, feat, ebuf, esum, ee, rst);
}
